// LinearAttention_17248588660972
// MI455X (gfx1250) — compile-verified, matrix operand order checked by reading, not run on hardware
//
#include <hip/hip_runtime.h>

// Problem constants (B=4, H=16, S=4096, D=64)
#define BH     64
#define SLEN   4096
#define DIM    64
#define NCHUNK 4
#define CHUNK  (SLEN / NCHUNK)   // 1024 rows of S per kv-block
#define TROWS  32                // S-rows per TDM tile (= one WMMA k-step)
#define OROWS  256               // rows of S per out-block

typedef __attribute__((ext_vector_type(16))) __bf16 v16bf;
typedef __attribute__((ext_vector_type(8)))  float  v8f;
typedef __attribute__((ext_vector_type(4)))  unsigned int v4u;
typedef __attribute__((ext_vector_type(8)))  int    v8i;
typedef __attribute__((ext_vector_type(4)))  int    v4i;

// elu(x)+1  ==  x>0 ? x+1 : exp(x)
__device__ __forceinline__ float featmap(float x) {
    return x > 0.0f ? x + 1.0f : __expf(x);
}

// K-index ordering inside A/B operand registers for 16-bit WMMA 16x16x32 (wave32):
// lanes 0-15  (half=0): k = 0..7 then 16..23
// lanes 16-31 (half=1): k = 8..15 then 24..31
__device__ __forceinline__ int kmap(int half, int i) {
    return (i < 8) ? (half * 8 + i) : (16 + half * 8 + (i - 8));
}

// ---- Tensor Data Mover: DMA a 2-D fp32 tile (tile0 x tile1) global -> LDS ----
// D# per CDNA5 ISA ch.8: group0 {count, lds_addr, global_addr, type=2},
// group1 {data_size, tensor dims/strides, tile dims}; groups 2/3 zero (<=2D).
// This toolchain's builtin is the 6-arg form:
//   (uint32x4 g0, int32x8 g1, int32x4 g2, int32x4 g3, int32x8 g4, i32 cpol)
__device__ __forceinline__ void tdm_load_2d(unsigned int lds_off, const void* gptr,
                                            unsigned int tdim0, unsigned int tdim1,
                                            unsigned int stride0,
                                            unsigned int tile0, unsigned int tile1) {
    const unsigned long long ga = (unsigned long long)(uintptr_t)gptr;
    v4u g0;
    g0[0] = 1u;                                      // count=1 valid descriptor
    g0[1] = lds_off;                                 // LDS byte address
    g0[2] = (unsigned int)(ga & 0xffffffffull);      // global_addr[31:0]
    g0[3] = (unsigned int)((ga >> 32) & 0x01fffffful) | (2u << 30);  // addr[56:32] | type=2
    v8i g1;
    g1[0] = (int)(2u << 16);                                         // data_size=4B
    g1[1] = (int)((tdim0 & 0xffffu) << 16);                          // tensor_dim0 lo
    g1[2] = (int)(((tdim0 >> 16) & 0xffffu) | ((tdim1 & 0xffffu) << 16));
    g1[3] = (int)(((tdim1 >> 16) & 0xffffu) | ((tile0 & 0xffffu) << 16));
    g1[4] = (int)(tile1 & 0xffffu);                                  // tile_dim1 (tile_dim2=0)
    g1[5] = (int)stride0;                                            // tensor_dim0_stride lo32
    g1[6] = 0;                                                       // stride0 hi / stride1 lo
    g1[7] = 0;
    const v4i z4 = {0, 0, 0, 0};
    const v8i z8 = {0, 0, 0, 0, 0, 0, 0, 0};
    __builtin_amdgcn_tensor_load_to_lds(g0, g1, z4, z4, z8, 0);
}

__device__ __forceinline__ unsigned int lds_addr_of(const void* p) {
    // generic LDS pointer: low 32 bits are the wave-relative LDS byte offset
    return (unsigned int)(uintptr_t)p;
}

__global__ void zero_ws_kernel(float* __restrict__ p, int n) {
    for (int i = blockIdx.x * blockDim.x + threadIdx.x; i < n;
         i += gridDim.x * blockDim.x)
        p[i] = 0.0f;
}

// Phase 1: KV[bh] += Kf^T V over a 1024-row S chunk; Ksum[bh] += column sums of Kf.
// 8 waves share each TDM-staged 32-row tile; wave w owns output tiles
// (mt = w/2, nt = 2*(w%2) + {0,1}) -> no cross-wave reduction needed.
__global__ __launch_bounds__(256)
void kv_kernel(const float* __restrict__ K, const float* __restrict__ V,
               const float* __restrict__ mask,
               float* __restrict__ KVg, float* __restrict__ Ksumg) {
    __shared__ float sK[2][TROWS * DIM];
    __shared__ float sV[2][TROWS * DIM];
    __shared__ float sM[2][TROWS];

    const int bh    = blockIdx.x / NCHUNK;
    const int chunk = blockIdx.x % NCHUNK;
    const int tid   = threadIdx.x;
    const int wave  = tid >> 5;
    const int lane  = tid & 31;
    const int half  = lane >> 4;
    const int lm    = lane & 15;

    const int mt  = wave >> 1;            // A (d) tile owned by this wave
    const int nt0 = (wave & 1) * 2;       // first of two B (e) tiles

    const float* __restrict__ Kp = K + (size_t)bh * SLEN * DIM;
    const float* __restrict__ Vp = V + (size_t)bh * SLEN * DIM;
    const float* __restrict__ Mp = mask + (size_t)bh * SLEN;

    v8f acc[2];
#pragma unroll
    for (int t = 0; t < 2; ++t)
#pragma unroll
        for (int r = 0; r < 8; ++r) acc[t][r] = 0.0f;
    float ksum_part = 0.0f;

    const int  s_base = chunk * CHUNK;
    const int  NT     = CHUNK / TROWS;    // 32 tiles per block

    if (wave == 0) {
        const int s0 = s_base;
        tdm_load_2d(lds_addr_of(&sK[0][0]), Kp + (size_t)s0 * DIM,
                    DIM, SLEN - s0, DIM, DIM, TROWS);
        tdm_load_2d(lds_addr_of(&sV[0][0]), Vp + (size_t)s0 * DIM,
                    DIM, SLEN - s0, DIM, DIM, TROWS);
        tdm_load_2d(lds_addr_of(&sM[0][0]), Mp + s0,
                    SLEN - s0, 1, SLEN, TROWS, 1);
    }

    for (int it = 0; it < NT; ++it) {
        const int buf = it & 1;
        if (wave == 0) __builtin_amdgcn_s_wait_tensorcnt(0);
        __syncthreads();   // publish tile `it`; all waves done with buf^1 (WAR)
        if (wave == 0 && it + 1 < NT) {
            const int s0 = s_base + (it + 1) * TROWS;
            tdm_load_2d(lds_addr_of(&sK[buf ^ 1][0]), Kp + (size_t)s0 * DIM,
                        DIM, SLEN - s0, DIM, DIM, TROWS);
            tdm_load_2d(lds_addr_of(&sV[buf ^ 1][0]), Vp + (size_t)s0 * DIM,
                        DIM, SLEN - s0, DIM, DIM, TROWS);
            tdm_load_2d(lds_addr_of(&sM[buf ^ 1][0]), Mp + s0,
                        SLEN - s0, 1, SLEN, TROWS, 1);
        }

        // Build operands from LDS tile and issue the wave's two WMMAs.
        v16bf A, B0, B1;
        float kp = 0.0f;
#pragma unroll
        for (int i = 0; i < 16; ++i) {
            const int rr = kmap(half, i);                 // row within tile
            const float mk = sM[buf][rr];
            const float f  = featmap(sK[buf][rr * DIM + mt * 16 + lm]) * mk;
            kp += f;
            A[i]  = (__bf16)f;
            B0[i] = (__bf16)sV[buf][rr * DIM + nt0 * 16 + lm];
            B1[i] = (__bf16)sV[buf][rr * DIM + (nt0 + 1) * 16 + lm];
        }
        if ((wave & 1) == 0) ksum_part += kp;   // avoid double count (A shared by 2 waves)

        acc[0] = __builtin_amdgcn_wmma_f32_16x16x32_bf16(
            false, A, false, B0, (short)0, acc[0], false, false);
        acc[1] = __builtin_amdgcn_wmma_f32_16x16x32_bf16(
            false, A, false, B1, (short)0, acc[1], false, false);
    }

    // Writeback: wave exclusively owns tiles (mt, nt0) and (mt, nt0+1).
    float* __restrict__ kvdst = KVg + (size_t)bh * DIM * DIM;
#pragma unroll
    for (int t = 0; t < 2; ++t)
#pragma unroll
        for (int r = 0; r < 8; ++r) {
            const int m = mt * 16 + r + 8 * half;   // C layout: vgpr r -> M=r / r+8
            const int n = (nt0 + t) * 16 + lm;      // N = lane % 16
            atomicAdd(&kvdst[m * DIM + n], acc[t][r]);
        }
    if ((wave & 1) == 0)
        atomicAdd(&Ksumg[bh * DIM + mt * 16 + lm], ksum_part);
}

// Phase 2: out = (Qf @ KV) * (1 / (Qf . Ksum))
__global__ __launch_bounds__(256)
void out_kernel(const float* __restrict__ Q, const float* __restrict__ KVg,
                const float* __restrict__ Ksumg, float* __restrict__ Out) {
    const int nsb  = SLEN / OROWS;               // 16 blocks per head
    const int bh   = blockIdx.x / nsb;
    const int sblk = blockIdx.x % nsb;
    const int tid  = threadIdx.x;
    const int wave = tid >> 5;
    const int lane = tid & 31;
    const int half = lane >> 4;
    const int lm   = lane & 15;

    __shared__ float ksum_l[DIM];
    if (tid < DIM) ksum_l[tid] = Ksumg[bh * DIM + tid];
    __syncthreads();

    // Preload KV as bf16 B tiles: B[ks][nt][k][n=lane] = KV[ks*32 + k][nt*16 + n]
    const float* __restrict__ KVp = KVg + (size_t)bh * DIM * DIM;
    v16bf Bkv[2][4];
#pragma unroll
    for (int ks = 0; ks < 2; ++ks)
#pragma unroll
        for (int i = 0; i < 16; ++i) {
            const int d = ks * 32 + kmap(half, i);
#pragma unroll
            for (int nt = 0; nt < 4; ++nt)
                Bkv[ks][nt][i] = (__bf16)KVp[d * DIM + nt * 16 + lm];
        }

    const float* __restrict__ Qp = Q + (size_t)bh * SLEN * DIM;
    float* __restrict__ Op       = Out + (size_t)bh * SLEN * DIM;

#pragma unroll
    for (int it = 0; it < 2; ++it) {             // 2 x 16 rows per wave
        const int srow0 = sblk * OROWS + wave * 32 + it * 16;
        const int s     = srow0 + lm;            // A-operand row for this lane

        // A tiles via b128 loads: per k-step the lane's d-indices are two
        // contiguous 8-float runs: [ks*32+half*8 .. +7] and [+16 .. +23].
        const float4* __restrict__ Qrow =
            (const float4*)(Qp + (size_t)s * DIM);
        float dp = 0.0f;
        v16bf A[2];
#pragma unroll
        for (int ks = 0; ks < 2; ++ks) {
            const int b4 = (ks * 32 + half * 8) >> 2;   // float4 index
            float qs[16];
            *(float4*)&qs[0]  = Qrow[b4 + 0];
            *(float4*)&qs[4]  = Qrow[b4 + 1];
            *(float4*)&qs[8]  = Qrow[b4 + 4];
            *(float4*)&qs[12] = Qrow[b4 + 5];
#pragma unroll
            for (int i = 0; i < 16; ++i) {
                const int d = ks * 32 + kmap(half, i);
                const float f = featmap(qs[i]);
                dp += f * ksum_l[d];
                A[ks][i] = (__bf16)f;
            }
        }
        // lanes L and L+16 hold complementary d-halves of the same row.
        const float denom = dp + __shfl_xor(dp, 16);

        v8f acc[4];
#pragma unroll
        for (int nt = 0; nt < 4; ++nt)
#pragma unroll
            for (int r = 0; r < 8; ++r) acc[nt][r] = 0.0f;

#pragma unroll
        for (int ks = 0; ks < 2; ++ks)
#pragma unroll
            for (int nt = 0; nt < 4; ++nt)
                acc[nt] = __builtin_amdgcn_wmma_f32_16x16x32_bf16(
                    false, A[ks], false, Bkv[ks][nt], (short)0, acc[nt],
                    false, false);

        // Scale by 1/denom(row) and store. C layout: vgpr r -> row r (lanes<16) / r+8.
#pragma unroll
        for (int r = 0; r < 8; ++r) {
            const float dn = __shfl(denom, r + 8 * half);
            const float z  = 1.0f / dn;
            const int  srw = srow0 + r + 8 * half;
            const size_t rowO = (size_t)srw * DIM;
#pragma unroll
            for (int nt = 0; nt < 4; ++nt)
                Op[rowO + nt * 16 + lm] = acc[nt][r] * z;
        }
    }
}

extern "C" void kernel_launch(void* const* d_in, const int* in_sizes, int n_in,
                              void* d_out, int out_size, void* d_ws, size_t ws_size,
                              hipStream_t stream) {
    (void)in_sizes; (void)n_in; (void)out_size; (void)ws_size;
    const float* Q    = (const float*)d_in[0];
    const float* K    = (const float*)d_in[1];
    const float* V    = (const float*)d_in[2];
    const float* mask = (const float*)d_in[3];
    float* Out = (float*)d_out;

    float* KVg   = (float*)d_ws;            // BH * 64 * 64
    float* Ksumg = KVg + BH * DIM * DIM;    // BH * 64
    const int nws = BH * DIM * DIM + BH * DIM;

    zero_ws_kernel<<<(nws + 255) / 256, 256, 0, stream>>>((float*)d_ws, nws);
    kv_kernel<<<BH * NCHUNK, 256, 0, stream>>>(K, V, mask, KVg, Ksumg);
    out_kernel<<<BH * (SLEN / OROWS), 256, 0, stream>>>(Q, KVg, Ksumg, Out);
}
